// UnsupervisedLoss_35416300323585
// MI455X (gfx1250) — matrix operand order repaired and checked by reading, not compiled, not run on hardware
//
#include <hip/hip_runtime.h>

// Problem constants (match reference: N=8192, D=512, K=20)
#define NN   8192
#define DD   512
#define KNEG 20
#define NCB  128   // number of 64-wide column blocks = NN/64

typedef __attribute__((ext_vector_type(16))) __bf16 v16bf;
typedef __attribute__((ext_vector_type(8)))  __bf16 v8bf;
typedef __attribute__((ext_vector_type(8)))  float  v8f;

union ABFrag { v16bf v; v8bf h[2]; };

// ---------------------------------------------------------------------------
// 1) Split X (fp32) into bf16 hi/lo parts: x = hi + lo (near-fp32 via 3 WMMAs)
// ---------------------------------------------------------------------------
__global__ __launch_bounds__(256)
void convert_split_kernel(const float* __restrict__ X,
                          unsigned short* __restrict__ XhiU,
                          unsigned short* __restrict__ XloU) {
    size_t i = (size_t)blockIdx.x * 256 + threadIdx.x;
    float x = X[i];
    __bf16 h = (__bf16)x;               // RNE f32 -> bf16
    __bf16 l = (__bf16)(x - (float)h);  // residual
    ((__bf16*)XhiU)[i] = h;
    ((__bf16*)XloU)[i] = l;
}

// ---------------------------------------------------------------------------
// 2) Fused gram (WMMA bf16 split) + log_sigmoid + adjacency mask.
//    Wave tile: 16 rows x 64 cols (4 accumulators). Block: 8 waves -> 128x64.
//    Grid: (NN/128, NN/64). Writes per-(colblock,row) partial sums of
//    ls*a and a (degree) -- each element written by exactly one wave.
// ---------------------------------------------------------------------------
__global__ __launch_bounds__(256)
void gram_pos_kernel(const unsigned short* __restrict__ XhiU,
                     const unsigned short* __restrict__ XloU,
                     const float* __restrict__ A,
                     float* __restrict__ posP,   // [NCB][NN]
                     float* __restrict__ degP)   // [NCB][NN]
{
    const int wave = threadIdx.x >> 5;
    const int lane = threadIdx.x & 31;
    const int lm   = lane & 15;     // position within 16-lane half
    const int lh   = lane >> 4;     // which half (0/1)

    const int rowBase = blockIdx.x * 128 + wave * 16;
    const int colBase = blockIdx.y * 64;

    v8f acc0 = {}; v8f acc1 = {}; v8f acc2 = {}; v8f acc3 = {};
    v8f acc[4] = {acc0, acc1, acc2, acc3};

    // A-operand (rows of gram): lane lm holds row rowBase+lm.
    // Per ISA layout: halves 0..7 -> K = k0 + 8*lh .. +7 ; 8..15 -> K = k0+16+8*lh .. +7
    const unsigned short* aHi = XhiU + (size_t)(rowBase + lm) * DD;
    const unsigned short* aLo = XloU + (size_t)(rowBase + lm) * DD;
    // B-operand (cols of gram = rows of X): same K layout as A: halves 0..7 -> K = k0 + 8*lh .. +7 ; 8..15 -> K = k0+16+8*lh .. +7
    const unsigned short* bHiBase = XhiU + (size_t)(colBase + lm) * DD;
    const unsigned short* bLoBase = XloU + (size_t)(colBase + lm) * DD;

    for (int k0 = 0; k0 < DD; k0 += 32) {
        ABFrag ahi, alo;
        ahi.h[0] = *(const v8bf*)(aHi + k0 + lh * 8);
        ahi.h[1] = *(const v8bf*)(aHi + k0 + 16 + lh * 8);
        alo.h[0] = *(const v8bf*)(aLo + k0 + lh * 8);
        alo.h[1] = *(const v8bf*)(aLo + k0 + 16 + lh * 8);
#pragma unroll
        for (int t = 0; t < 4; ++t) {
            const unsigned short* bh = bHiBase + (size_t)(t * 16) * DD + k0 + lh * 8;
            const unsigned short* bl = bLoBase + (size_t)(t * 16) * DD + k0 + lh * 8;
            ABFrag bhi, blo;
            bhi.h[0] = *(const v8bf*)(bh);
            bhi.h[1] = *(const v8bf*)(bh + 16);
            blo.h[0] = *(const v8bf*)(bl);
            blo.h[1] = *(const v8bf*)(bl + 16);
            // split-bf16: hi*hi + hi*lo + lo*hi
            acc[t] = __builtin_amdgcn_wmma_f32_16x16x32_bf16(false, ahi.v, false, bhi.v,
                                                             (short)0, acc[t], false, false);
            acc[t] = __builtin_amdgcn_wmma_f32_16x16x32_bf16(false, ahi.v, false, blo.v,
                                                             (short)0, acc[t], false, false);
            acc[t] = __builtin_amdgcn_wmma_f32_16x16x32_bf16(false, alo.v, false, bhi.v,
                                                             (short)0, acc[t], false, false);
        }
    }

    // Epilogue: C/D layout -> element (r, lane): row = rowBase + r + 8*lh, col = colBase+16t+lm
    float posAcc[8], degAcc[8];
#pragma unroll
    for (int r = 0; r < 8; ++r) { posAcc[r] = 0.f; degAcc[r] = 0.f; }

#pragma unroll
    for (int t = 0; t < 4; ++t) {
        const int col = colBase + t * 16 + lm;
#pragma unroll
        for (int r = 0; r < 8; ++r) {
            const int row = rowBase + r + 8 * lh;
            // Stream adjacency non-temporally: 256 MB single-use, keep L2 for X.
            float a = __builtin_nontemporal_load(&A[(size_t)row * NN + col]);
            float s = acc[t][r];
            // log_sigmoid(s) = min(s,0) - log1p(exp(-|s|))
            float ls = fminf(s, 0.f) - __logf(1.f + __expf(-fabsf(s)));
            posAcc[r] += ls * a;
            degAcc[r] += a;
        }
    }

    // Reduce across the 16 lanes of each half (same row, different cols).
#pragma unroll
    for (int r = 0; r < 8; ++r) {
        float p  = posAcc[r];
        float dg = degAcc[r];
#pragma unroll
        for (int off = 1; off < 16; off <<= 1) {
            p  += __shfl_xor(p,  off, 32);
            dg += __shfl_xor(dg, off, 32);
        }
        if (lm == 0) {
            const int row = rowBase + r + 8 * lh;
            posP[(size_t)blockIdx.y * NN + row] = p;
            degP[(size_t)blockIdx.y * NN + row] = dg;
        }
    }
}

// ---------------------------------------------------------------------------
// 3) Negative sampling term: one block per node. x[n] staged in LDS,
//    20 dot products in fp32 (rows are L2-resident).
// ---------------------------------------------------------------------------
__global__ __launch_bounds__(256)
void neg_kernel(const float* __restrict__ X,
                const int* __restrict__ negIdx,
                float* __restrict__ negVal) {
    __shared__ float xs[DD];
    __shared__ float wsum[8];
    const int n = blockIdx.x;
    for (int i = threadIdx.x; i < DD; i += 256) xs[i] = X[(size_t)n * DD + i];
    __syncthreads();

    const int wave = threadIdx.x >> 5;
    const int lane = threadIdx.x & 31;
    float acc = 0.f;
    for (int k = wave; k < KNEG; k += 8) {
        const int j = negIdx[n * KNEG + k];
        const float* xj = X + (size_t)j * DD;
        float dot = 0.f;
        for (int d = lane; d < DD; d += 32) dot += xs[d] * xj[d];
#pragma unroll
        for (int off = 16; off; off >>= 1) dot += __shfl_xor(dot, off, 32);
        // -log_sigmoid(-s) = softplus(s) = max(s,0) + log1p(exp(-|s|))
        acc += fmaxf(dot, 0.f) + __logf(1.f + __expf(-fabsf(dot)));
    }
    if (lane == 0) wsum[wave] = acc;
    __syncthreads();
    if (threadIdx.x == 0) {
        float s = 0.f;
#pragma unroll
        for (int w = 0; w < 8; ++w) s += wsum[w];
        negVal[n] = s * (1.0f / KNEG);
    }
}

// ---------------------------------------------------------------------------
// 4) Deterministic two-stage reduction to the scalar loss.
// ---------------------------------------------------------------------------
__global__ __launch_bounds__(256)
void reduce_kernel(const float* __restrict__ posP,
                   const float* __restrict__ degP,
                   const float* __restrict__ negVal,
                   float* __restrict__ blockSums) {
    const int n = blockIdx.x * 256 + threadIdx.x;
    float p = 0.f, dg = 0.f;
    for (int cb = 0; cb < NCB; ++cb) {
        p  += posP[(size_t)cb * NN + n];
        dg += degP[(size_t)cb * NN + n];
    }
    float val = -p / dg + negVal[n];
    __shared__ float sh[256];
    sh[threadIdx.x] = val;
    __syncthreads();
    for (int s = 128; s; s >>= 1) {
        if (threadIdx.x < s) sh[threadIdx.x] += sh[threadIdx.x + s];
        __syncthreads();
    }
    if (threadIdx.x == 0) blockSums[blockIdx.x] = sh[0];
}

__global__ void final_kernel(const float* __restrict__ blockSums, float* __restrict__ out) {
    float v = blockSums[threadIdx.x];  // exactly 32 block sums
#pragma unroll
    for (int off = 16; off; off >>= 1) v += __shfl_xor(v, off, 32);
    if (threadIdx.x == 0) out[0] = v;
}

// ---------------------------------------------------------------------------
extern "C" void kernel_launch(void* const* d_in, const int* in_sizes, int n_in,
                              void* d_out, int out_size, void* d_ws, size_t ws_size,
                              hipStream_t stream) {
    const float* X      = (const float*)d_in[0];  // [1,N,D] fp32
    const float* A      = (const float*)d_in[1];  // [1,N,N] fp32
    const int*   negIdx = (const int*)d_in[2];    // [N,K] int32
    float*       out    = (float*)d_out;          // scalar

    char* ws = (char*)d_ws;
    unsigned short* Xhi = (unsigned short*)ws;                           // 8 MB
    unsigned short* Xlo = (unsigned short*)(ws + (size_t)NN * DD * 2);   // 8 MB
    float* posP      = (float*)(ws + (size_t)NN * DD * 4);               // 4 MB
    float* degP      = posP + (size_t)NCB * NN;                          // 4 MB
    float* negVal    = degP + (size_t)NCB * NN;                          // 32 KB
    float* blockSums = negVal + NN;                                      // 128 B

    convert_split_kernel<<<(NN * DD) / 256, 256, 0, stream>>>(X, Xhi, Xlo);

    dim3 grid(NN / 128, NN / 64);
    gram_pos_kernel<<<grid, 256, 0, stream>>>(Xhi, Xlo, A, posP, degP);

    neg_kernel<<<NN, 256, 0, stream>>>(X, negIdx, negVal);

    reduce_kernel<<<NN / 256, 256, 0, stream>>>(posP, degP, negVal, blockSums);
    final_kernel<<<1, 32, 0, stream>>>(blockSums, out);
}
